// RoPEMultiheadAttention_47648367182425
// MI455X (gfx1250) — matrix operand order repaired and checked by reading, not compiled, not run on hardware
//
#include <hip/hip_runtime.h>
#include <hip/hip_bf16.h>

// ---------------- types ----------------
typedef __attribute__((ext_vector_type(16))) __bf16 v16bf;
typedef __attribute__((ext_vector_type(8)))  float  v8f;

union Frag16 { v16bf v; uint4 q[2]; };

// ---------------- bf16 helpers (pure bit ops) ----------------
__device__ __forceinline__ unsigned short f2bf(float f) {
    union { float f; unsigned u; } x; x.f = f;
    unsigned r = x.u + 0x7FFFu + ((x.u >> 16) & 1u);
    return (unsigned short)(r >> 16);
}
__device__ __forceinline__ float bf2f(unsigned short h) {
    union { unsigned u; float f; } x; x.u = ((unsigned)h) << 16;
    return x.f;
}

// ---------------- WMMA fragment loaders from LDS ----------------
// A fragment (16x32 bf16, MxK): lane l -> row M=l%16.
//   VGPR0..3: K = 8*(l/16)+{0..7};  VGPR4..7: K = 16+8*(l/16)+{0..7}
__device__ __forceinline__ v16bf lds_frag_A(const unsigned short* p, int r0, int stride, int k0) {
    int lane = threadIdx.x & 31;
    int half = lane >> 4, m = lane & 15;
    const unsigned short* row = p + (size_t)(r0 + m) * stride;
    Frag16 f;
    f.q[0] = *(const uint4*)(row + k0 + 8 * half);
    f.q[1] = *(const uint4*)(row + k0 + 16 + 8 * half);
    return f.v;
}
// B fragment (32x16 bf16, KxN) sourced from an LDS tile stored [N][K]:
//   lane l -> col N=l%16, VGPR0..3: K = 8*(l/16)+{0..7};  VGPR4..7: K = 16+8*(l/16)+{0..7}
__device__ __forceinline__ v16bf lds_frag_B(const unsigned short* p, int n0, int stride, int k0) {
    int lane = threadIdx.x & 31;
    int half = lane >> 4, n = lane & 15;
    const unsigned short* row = p + (size_t)(n0 + n) * stride;
    Frag16 f;
    f.q[0] = *(const uint4*)(row + k0 + 8 * half);
    f.q[1] = *(const uint4*)(row + k0 + 16 + 8 * half);
    return f.v;
}

#define WMMA_BF16(A, B, C) __builtin_amdgcn_wmma_f32_16x16x32_bf16(false, A, false, B, (short)0, C, false, false)

// ---------------- problem dims ----------------
#define BB 4
#define LL 2048
#define EE 1024
#define HH 16
#define DD 64
#define NQKV 3072
#define MTOT (BB * LL) // 8192
#define GSTR 72        // padded LDS row stride (elements)

// ---------------- kernel 1a: f32 -> bf16 convert ----------------
__global__ void cvt_f32_bf16(const float* __restrict__ src, unsigned short* __restrict__ dst, int n) {
    int i = blockIdx.x * blockDim.x + threadIdx.x;
    if (i < n) dst[i] = f2bf(src[i]);
}

// ---------------- kernel 1b: f32 -> bf16 convert + transpose ----------------
// src [rows][cols] f32  ->  dst [cols][rows] bf16   (LDS-tiled 32x32)
__global__ __launch_bounds__(256) void cvt_transpose(const float* __restrict__ src,
                                                     unsigned short* __restrict__ dst,
                                                     int rows, int cols) {
    __shared__ unsigned short t[32][33];
    int r0 = blockIdx.y * 32, c0 = blockIdx.x * 32;
    int tr = threadIdx.x & 31, tc = threadIdx.x >> 5;
#pragma unroll
    for (int i = 0; i < 32; i += 8)
        t[tc + i][tr] = f2bf(src[(size_t)(r0 + tc + i) * cols + c0 + tr]);
    __syncthreads();
#pragma unroll
    for (int i = 0; i < 32; i += 8)
        dst[(size_t)(c0 + tc + i) * rows + r0 + tr] = t[tr][tc + i];
}

// ---------------- kernel 2: QKV GEMM (bf16 WMMA), scatter to per-head ----------------
// X [8192,1024] bf16, Wt [3072,1024] bf16 (pre-transposed) -> Q,K: [B,H,L,D]; V: [B,H,D,L]
__global__ __launch_bounds__(256) void gemm_qkv(
    const unsigned short* __restrict__ X,
    const unsigned short* __restrict__ Wt,
    const float* __restrict__ bias,
    unsigned short* __restrict__ Qb,
    unsigned short* __restrict__ Kb,
    unsigned short* __restrict__ Vb)
{
    __shared__ __align__(16) unsigned short As[128 * GSTR];
    __shared__ __align__(16) unsigned short Bs[128 * GSTR];

    int tid = threadIdx.x, lane = tid & 31, wid = tid >> 5;
    int bm = blockIdx.y * 128, bn = blockIdx.x * 128;
    int wm = (wid >> 1) * 32;  // 4 waves in M
    int wn = (wid & 1) * 64;   // 2 waves in N

    v8f acc[2][4] = {};

    for (int kt = 0; kt < EE; kt += 64) {
        __syncthreads();
#pragma unroll
        for (int i = 0; i < 4; ++i) { // A tile 128x64, coalesced b128 copy
            int c = tid + i * 256;
            int r = c >> 3, off = (c & 7) * 8;
            *(uint4*)(As + r * GSTR + off) = *(const uint4*)(X + (size_t)(bm + r) * EE + kt + off);
        }
#pragma unroll
        for (int i = 0; i < 4; ++i) { // B tile 128(N)x64(K), plain copy (W pre-transposed)
            int c = tid + i * 256;
            int r = c >> 3, off = (c & 7) * 8;
            const unsigned short* g = Wt + (size_t)(bn + r) * EE + kt + off;
            *(uint4*)(Bs + r * GSTR + off) = *(const uint4*)g;
            __builtin_prefetch(g + 64, 0, 1);
        }
        __syncthreads();

#pragma unroll
        for (int ks = 0; ks < 2; ++ks) {
            v16bf a0 = lds_frag_A(As, wm,      GSTR, ks * 32);
            v16bf a1 = lds_frag_A(As, wm + 16, GSTR, ks * 32);
            v16bf b0 = lds_frag_B(Bs, wn,      GSTR, ks * 32);
            v16bf b1 = lds_frag_B(Bs, wn + 16, GSTR, ks * 32);
            v16bf b2 = lds_frag_B(Bs, wn + 32, GSTR, ks * 32);
            v16bf b3 = lds_frag_B(Bs, wn + 48, GSTR, ks * 32);
            acc[0][0] = WMMA_BF16(a0, b0, acc[0][0]);
            acc[0][1] = WMMA_BF16(a0, b1, acc[0][1]);
            acc[0][2] = WMMA_BF16(a0, b2, acc[0][2]);
            acc[0][3] = WMMA_BF16(a0, b3, acc[0][3]);
            acc[1][0] = WMMA_BF16(a1, b0, acc[1][0]);
            acc[1][1] = WMMA_BF16(a1, b1, acc[1][1]);
            acc[1][2] = WMMA_BF16(a1, b2, acc[1][2]);
            acc[1][3] = WMMA_BF16(a1, b3, acc[1][3]);
        }
    }

    // Epilogue: s (q/k/v), h, and b are wave-uniform per n-tile -> scalar branch,
    // per-lane work is a single strided bf16 store stream per tile.
    int half = lane >> 4, ln = lane & 15;
    int bq = bm >> 11;                  // uniform batch index (tiles never straddle)
    int lbase = (bm & 2047) + wm;       // uniform local-sequence base
#pragma unroll
    for (int ni = 0; ni < 4; ++ni) {
        int n0 = bn + wn + ni * 16;     // uniform, 16-aligned
        int s = n0 >> 10;               // uniform: 0=q 1=k 2=v
        int rem = n0 & 1023;
        int h = rem >> 6, d0 = rem & 63; // uniform
        size_t bh = (size_t)bq * HH + h;
        float bv = bias[n0 + ln];
        if (s < 2) {
            unsigned short* base = (s == 0 ? Qb : Kb) + bh * LL * DD + d0 + ln;
#pragma unroll
            for (int mi = 0; mi < 2; ++mi)
#pragma unroll
                for (int r = 0; r < 8; ++r) {
                    int l = lbase + mi * 16 + r + 8 * half;
                    base[(size_t)l * DD] = f2bf(acc[mi][ni][r] + bv);
                }
        } else {
            unsigned short* base = Vb + (bh * DD + d0 + ln) * LL; // V stored transposed
#pragma unroll
            for (int mi = 0; mi < 2; ++mi)
#pragma unroll
                for (int r = 0; r < 8; ++r) {
                    int l = lbase + mi * 16 + r + 8 * half;
                    base[l] = f2bf(acc[mi][ni][r] + bv);
                }
        }
    }
}

// ---------------- kernel 3: RoPE on Q,K in [B,H,L,D] bf16 ----------------
__global__ void rope_kernel(unsigned short* __restrict__ Qb,
                            unsigned short* __restrict__ Kb,
                            const float* __restrict__ freqs) // [L,D]
{
    int i = blockIdx.x * blockDim.x + threadIdx.x;
    const int total = BB * HH * LL * (DD / 2);
    if (i >= total) return;
    int j = i & 31;
    int row = i >> 5;          // (b*H+h)*L + l
    int l = row & (LL - 1);
    float f1 = freqs[l * DD + j];
    float f2 = freqs[l * DD + j + 32];
    float c1 = __cosf(f1), s1 = __sinf(f1);
    float c2 = __cosf(f2), s2 = __sinf(f2);
    size_t base = (size_t)row * DD;

    float q1 = bf2f(Qb[base + j]), q2 = bf2f(Qb[base + j + 32]);
    Qb[base + j]      = f2bf(q1 * c1 - q2 * s1);
    Qb[base + j + 32] = f2bf(q2 * c2 + q1 * s2);

    float k1 = bf2f(Kb[base + j]), k2 = bf2f(Kb[base + j + 32]);
    Kb[base + j]      = f2bf(k1 * c1 - k2 * s1);
    Kb[base + j + 32] = f2bf(k2 * c2 + k1 * s2);
}

// ---------------- kernel 4: flash attention per (b,h), 128 q-rows/block ----------------
__global__ __launch_bounds__(256) void attn_kernel(
    const unsigned short* __restrict__ Qb,   // [B,H,L,D]
    const unsigned short* __restrict__ Kb,   // [B,H,L,D]
    const unsigned short* __restrict__ Vb,   // [B,H,D,L] (transposed)
    unsigned short* __restrict__ Ob)         // [B,L,E]
{
    __shared__ __align__(16) unsigned short Qs[128 * GSTR];
    __shared__ __align__(16) unsigned short KsS[64 * GSTR]; // [key][d] == Bt for S
    __shared__ __align__(16) unsigned short Vt[64 * GSTR];  // [d][key] == Bt for O
    __shared__ __align__(16) unsigned short Ps[8 * 16 * GSTR];

    int tid = threadIdx.x, lane = tid & 31, wid = tid >> 5;
    int q0 = blockIdx.x * 128;
    int bh = blockIdx.y;
    const unsigned short* Qg = Qb + (size_t)bh * LL * DD;
    const unsigned short* Kg = Kb + (size_t)bh * LL * DD;
    const unsigned short* Vg = Vb + (size_t)bh * DD * LL;

#pragma unroll
    for (int i = 0; i < 4; ++i) { // Q tile 128x64
        int c = tid + i * 256;
        int r = c >> 3, off = (c & 7) * 8;
        *(uint4*)(Qs + r * GSTR + off) = *(const uint4*)(Qg + (size_t)(q0 + r) * DD + off);
    }

    float rm[8], rl[8];
    v8f o[4] = {};
#pragma unroll
    for (int r = 0; r < 8; ++r) { rm[r] = -1e30f; rl[r] = 0.f; }

    unsigned short* Pw = Ps + wid * 16 * GSTR;
    int half = lane >> 4, ln = lane & 15;

    for (int kb = 0; kb < LL; kb += 64) {
        __syncthreads();
#pragma unroll
        for (int i = 0; i < 2; ++i) { // K tile 64x64
            int c = tid + i * 256;
            int r = c >> 3, off = (c & 7) * 8;
            *(uint4*)(KsS + r * GSTR + off) = *(const uint4*)(Kg + (size_t)(kb + r) * DD + off);
        }
#pragma unroll
        for (int i = 0; i < 2; ++i) { // V^T tile 64(d)x64(key), plain copy
            int c = tid + i * 256;
            int r = c >> 3, off = (c & 7) * 8;
            *(uint4*)(Vt + r * GSTR + off) = *(const uint4*)(Vg + (size_t)r * LL + kb + off);
        }
        __syncthreads();

        // S = Q * K^T
        v8f s[4] = {};
#pragma unroll
        for (int ks = 0; ks < 2; ++ks) {
            v16bf a  = lds_frag_A(Qs, wid * 16, GSTR, ks * 32);
            v16bf b0 = lds_frag_B(KsS, 0,  GSTR, ks * 32);
            v16bf b1 = lds_frag_B(KsS, 16, GSTR, ks * 32);
            v16bf b2 = lds_frag_B(KsS, 32, GSTR, ks * 32);
            v16bf b3 = lds_frag_B(KsS, 48, GSTR, ks * 32);
            s[0] = WMMA_BF16(a, b0, s[0]);
            s[1] = WMMA_BF16(a, b1, s[1]);
            s[2] = WMMA_BF16(a, b2, s[2]);
            s[3] = WMMA_BF16(a, b3, s[3]);
        }
        // online softmax (scale 1/sqrt(64) = 0.125)
#pragma unroll
        for (int r = 0; r < 8; ++r) {
            float mx = rm[r];
#pragma unroll
            for (int ni = 0; ni < 4; ++ni) mx = fmaxf(mx, s[ni][r] * 0.125f);
#pragma unroll
            for (int off = 1; off < 16; off <<= 1) mx = fmaxf(mx, __shfl_xor(mx, off, 32));
            float sc = __expf(rm[r] - mx);
            rm[r] = mx;
            float rs = 0.f;
#pragma unroll
            for (int ni = 0; ni < 4; ++ni) {
                float p = __expf(s[ni][r] * 0.125f - mx);
                s[ni][r] = p;
                rs += p;
            }
#pragma unroll
            for (int off = 1; off < 16; off <<= 1) rs += __shfl_xor(rs, off, 32);
            rl[r] = rl[r] * sc + rs;
#pragma unroll
            for (int ni = 0; ni < 4; ++ni) o[ni][r] *= sc;
        }
        // P (C-layout) -> wave-private LDS -> A-fragments
#pragma unroll
        for (int ni = 0; ni < 4; ++ni)
#pragma unroll
            for (int r = 0; r < 8; ++r)
                Pw[(r + 8 * half) * GSTR + ni * 16 + ln] = f2bf(s[ni][r]);
        // O += P * V
#pragma unroll
        for (int ks = 0; ks < 2; ++ks) {
            v16bf a  = lds_frag_A(Pw, 0, GSTR, ks * 32);
            v16bf b0 = lds_frag_B(Vt, 0,  GSTR, ks * 32);
            v16bf b1 = lds_frag_B(Vt, 16, GSTR, ks * 32);
            v16bf b2 = lds_frag_B(Vt, 32, GSTR, ks * 32);
            v16bf b3 = lds_frag_B(Vt, 48, GSTR, ks * 32);
            o[0] = WMMA_BF16(a, b0, o[0]);
            o[1] = WMMA_BF16(a, b1, o[1]);
            o[2] = WMMA_BF16(a, b2, o[2]);
            o[3] = WMMA_BF16(a, b3, o[3]);
        }
    }

    int b = bh >> 4, h = bh & 15;
#pragma unroll
    for (int r = 0; r < 8; ++r) {
        float inv = 1.0f / rl[r];
        int qrow = q0 + wid * 16 + r + 8 * half;
#pragma unroll
        for (int ni = 0; ni < 4; ++ni) {
            int col = h * DD + ni * 16 + ln;
            Ob[((size_t)b * LL + qrow) * EE + col] = f2bf(o[ni][r] * inv);
        }
    }
}

// ---------------- kernel 5: output GEMM (bf16 WMMA) + bias -> f32 ----------------
__global__ __launch_bounds__(256) void gemm_out(
    const unsigned short* __restrict__ A,    // [8192,1024] bf16
    const unsigned short* __restrict__ Wt,   // [1024,1024] bf16 (pre-transposed)
    const float* __restrict__ bias,
    float* __restrict__ out)                 // [8192,1024] f32
{
    __shared__ __align__(16) unsigned short As[128 * GSTR];
    __shared__ __align__(16) unsigned short Bs[128 * GSTR];

    int tid = threadIdx.x, lane = tid & 31, wid = tid >> 5;
    int bm = blockIdx.y * 128, bn = blockIdx.x * 128;
    int wm = (wid >> 1) * 32;
    int wn = (wid & 1) * 64;

    v8f acc[2][4] = {};

    for (int kt = 0; kt < EE; kt += 64) {
        __syncthreads();
#pragma unroll
        for (int i = 0; i < 4; ++i) {
            int c = tid + i * 256;
            int r = c >> 3, off = (c & 7) * 8;
            *(uint4*)(As + r * GSTR + off) = *(const uint4*)(A + (size_t)(bm + r) * EE + kt + off);
        }
#pragma unroll
        for (int i = 0; i < 4; ++i) {
            int c = tid + i * 256;
            int r = c >> 3, off = (c & 7) * 8;
            const unsigned short* g = Wt + (size_t)(bn + r) * EE + kt + off;
            *(uint4*)(Bs + r * GSTR + off) = *(const uint4*)g;
            __builtin_prefetch(g + 64, 0, 1);
        }
        __syncthreads();

#pragma unroll
        for (int ks = 0; ks < 2; ++ks) {
            v16bf a0 = lds_frag_A(As, wm,      GSTR, ks * 32);
            v16bf a1 = lds_frag_A(As, wm + 16, GSTR, ks * 32);
            v16bf b0 = lds_frag_B(Bs, wn,      GSTR, ks * 32);
            v16bf b1 = lds_frag_B(Bs, wn + 16, GSTR, ks * 32);
            v16bf b2 = lds_frag_B(Bs, wn + 32, GSTR, ks * 32);
            v16bf b3 = lds_frag_B(Bs, wn + 48, GSTR, ks * 32);
            acc[0][0] = WMMA_BF16(a0, b0, acc[0][0]);
            acc[0][1] = WMMA_BF16(a0, b1, acc[0][1]);
            acc[0][2] = WMMA_BF16(a0, b2, acc[0][2]);
            acc[0][3] = WMMA_BF16(a0, b3, acc[0][3]);
            acc[1][0] = WMMA_BF16(a1, b0, acc[1][0]);
            acc[1][1] = WMMA_BF16(a1, b1, acc[1][1]);
            acc[1][2] = WMMA_BF16(a1, b2, acc[1][2]);
            acc[1][3] = WMMA_BF16(a1, b3, acc[1][3]);
        }
    }

    int half = lane >> 4, ln = lane & 15;
#pragma unroll
    for (int ni = 0; ni < 4; ++ni) {
        int n = bn + wn + ni * 16 + ln;
        float bv = bias[n];
        float* base = out + n;
#pragma unroll
        for (int mi = 0; mi < 2; ++mi)
#pragma unroll
            for (int r = 0; r < 8; ++r) {
                int m = bm + wm + mi * 16 + r + 8 * half;
                base[(size_t)m * EE] = acc[mi][ni][r] + bv;
            }
    }
}

// ---------------- host launcher ----------------
extern "C" void kernel_launch(void* const* d_in, const int* in_sizes, int n_in,
                              void* d_out, int out_size, void* d_ws, size_t ws_size,
                              hipStream_t stream) {
    const float* x     = (const float*)d_in[0]; // [B,L,E]
    const float* freqs = (const float*)d_in[1]; // [L,D]
    const float* Wqkv  = (const float*)d_in[2]; // [E,3E]
    const float* bqkv  = (const float*)d_in[3]; // [3E]
    const float* Wout  = (const float*)d_in[4]; // [E,E]
    const float* bout  = (const float*)d_in[5]; // [E]
    float* out = (float*)d_out;

    unsigned short* ws = (unsigned short*)d_ws;
    const size_t nX    = (size_t)MTOT * EE;
    const size_t nWqkv = (size_t)EE * NQKV;
    const size_t nWout = (size_t)EE * EE;
    const size_t nHead = (size_t)BB * HH * LL * DD;

    unsigned short* Xbf  = ws;
    unsigned short* Wqt  = Xbf + nX;      // [3E,E] transposed bf16
    unsigned short* Wot  = Wqt + nWqkv;   // [E,E]  transposed bf16
    unsigned short* Qb   = Wot + nWout;   // [B,H,L,D]
    unsigned short* Kb   = Qb + nHead;    // [B,H,L,D]
    unsigned short* Vb   = Kb + nHead;    // [B,H,D,L]
    unsigned short* Ob   = Vb + nHead;    // [B,L,E]

    cvt_f32_bf16<<<(int)((nX + 255) / 256), 256, 0, stream>>>(x, Xbf, (int)nX);
    cvt_transpose<<<dim3(NQKV / 32, EE / 32), 256, 0, stream>>>(Wqkv, Wqt, EE, NQKV);
    cvt_transpose<<<dim3(EE / 32, EE / 32), 256, 0, stream>>>(Wout, Wot, EE, EE);

    gemm_qkv<<<dim3(NQKV / 128, MTOT / 128), 256, 0, stream>>>(Xbf, Wqt, bqkv, Qb, Kb, Vb);

    {
        int total = BB * HH * LL * (DD / 2);
        rope_kernel<<<(total + 255) / 256, 256, 0, stream>>>(Qb, Kb, freqs);
    }

    attn_kernel<<<dim3(LL / 128, BB * HH), 256, 0, stream>>>(Qb, Kb, Vb, Ob);

    gemm_out<<<dim3(EE / 128, MTOT / 128), 256, 0, stream>>>(Ob, Wot, bout, out);
}
